// SpikeRNN_60481729462984
// MI455X (gfx1250) — matrix operand order repaired and checked by reading, not compiled, not run on hardware
//
#include <hip/hip_runtime.h>
#include <hip/hip_bf16.h>
#include <math.h>

// Problem dims (from reference setup_inputs)
constexpr int BB = 512;
constexpr int TT = 50;
constexpr int DD = 1156;
constexpr int HH = 1500;
constexpr int KP_IN = 1184;   // D padded to multiple of 32
constexpr int KP_H  = 1504;   // H padded to multiple of 32
constexpr int NP    = 1536;   // H padded to multiple of 64 (weight-row padding)

typedef __attribute__((ext_vector_type(16))) _Float16 v16h;
typedef __attribute__((ext_vector_type(8)))  float    v8f;

union FragH { v16h v; unsigned u[8]; };
union FragF { v8f  v; float    f[8]; };

// Async global->LDS 128-bit copy (CDNA5 GLOBAL_LOAD_ASYNC_TO_LDS_B128,
// tracked by ASYNCcnt; no VGPR data path).
__device__ __forceinline__ void async_copy_b128(void* lds_ptr, const void* gptr) {
  unsigned lds_off = (unsigned)(uintptr_t)lds_ptr;  // LDS aperture: addr[31:0] = LDS byte offset
  asm volatile("global_load_async_to_lds_b128 %0, %1, off"
               :: "v"(lds_off), "v"(gptr) : "memory");
}

// ---------------------------------------------------------------------------
// f16 WMMA GEMM with async-LDS double-buffered pipeline:
//   C[M,Nv] = A[M,K] * W[NP,K]^T          (K % 32 == 0, NP % 64 == 0, NP >= Nv)
// Block tile 128(M) x 64(N), 256 threads = 8 waves, wave tile 32x32 (2x2 WMMA)
// ---------------------------------------------------------------------------
__global__ __launch_bounds__(256) void wmma_gemm_f16(
    const _Float16* __restrict__ A,   // [M, K] row-major, fully padded
    const _Float16* __restrict__ W,   // [NP, K] row-major, rows >= Nv zero-padded
    float* __restrict__ C,            // [M, Nv]
    int M, int Nv, int K)
{
  __shared__ alignas(16) _Float16 As[2][128 * 40];
  __shared__ alignas(16) _Float16 Bs[2][64 * 40];

  const int tid  = threadIdx.x;
  const int lane = tid & 31;          // wave32
  const int wv   = tid >> 5;
  const int wr   = wv & 3;            // wave M sub-tile (0..3) * 32
  const int wc   = wv >> 2;           // wave N sub-tile (0..1) * 32
  const int m0   = blockIdx.y * 128;
  const int n0   = blockIdx.x * 64;
  const int lh   = lane & 15;
  const bool hi  = lane >= 16;

  // Per-thread staging coordinates (3 x b128 async ops per thread per tile)
  const int ra0 = tid >> 2,          sa0 = tid & 3;          // A chunk 0 (rows 0..63)
  const int ra1 = (tid + 256) >> 2,  sa1 = tid & 3;          // A chunk 1 (rows 64..127)
  const int rb  = tid >> 2,          sb  = tid & 3;          // B chunk  (rows 0..63)

  FragF acc[2][2];
  #pragma unroll
  for (int i = 0; i < 2; ++i)
    #pragma unroll
    for (int j = 0; j < 2; ++j)
      #pragma unroll
      for (int e = 0; e < 8; ++e) acc[i][j].f[e] = 0.0f;

  // Issue one K-tile's worth of async copies into LDS buffer bf
  auto issue_tile = [&](int k0, int bf) {
    async_copy_b128(&As[bf][ra0 * 40 + sa0 * 8],
                    A + (size_t)(m0 + ra0) * K + (k0 + sa0 * 8));
    async_copy_b128(&As[bf][ra1 * 40 + sa1 * 8],
                    A + (size_t)(m0 + ra1) * K + (k0 + sa1 * 8));
    async_copy_b128(&Bs[bf][rb * 40 + sb * 8],
                    W + (size_t)(n0 + rb) * K + (k0 + sb * 8));
  };

  const int nt = K >> 5;              // K tiles of 32
  issue_tile(0, 0);

  for (int kt = 0; kt < nt; ++kt) {
    const int bf = kt & 1;
    if (kt + 1 < nt) {
      issue_tile((kt + 1) << 5, bf ^ 1);
      // Loads complete in order: asynccnt<=3 => current tile resident,
      // next tile still streaming (copy/compute overlap).
      asm volatile("s_wait_asynccnt 0x3" ::: "memory");
    } else {
      asm volatile("s_wait_asynccnt 0x0" ::: "memory");
    }
    __syncthreads();                  // all waves' tile-kt data visible in LDS

    // Build fragments per ISA 7.12.2 layouts (16-bit A 16x32, B 32x16)
    FragH aF[2], bF[2];
    #pragma unroll
    for (int mt = 0; mt < 2; ++mt) {
      const _Float16* ap = &As[bf][(wr * 32 + mt * 16 + lh) * 40];
      const int kb = hi ? 8 : 0;      // lanes 16-31 hold K+8 / K+24 halves
      #pragma unroll
      for (int i = 0; i < 8; ++i) {
        int k = kb + ((i < 4) ? (2 * i) : (2 * i + 8));
        aF[mt].u[i] = *reinterpret_cast<const unsigned*>(ap + k);
      }
    }
    #pragma unroll
    for (int nt2 = 0; nt2 < 2; ++nt2) {
      const _Float16* bp = &Bs[bf][(wc * 32 + nt2 * 16 + lh) * 40];
      #pragma unroll
      for (int i = 0; i < 8; ++i)
        bF[nt2].u[i] = *reinterpret_cast<const unsigned*>(bp + (hi ? 8 : 0) + ((i < 4) ? (2 * i) : (2 * i + 8)));
    }

    #pragma unroll
    for (int mt = 0; mt < 2; ++mt)
      #pragma unroll
      for (int nt2 = 0; nt2 < 2; ++nt2)
        acc[mt][nt2].v = __builtin_amdgcn_wmma_f32_16x16x32_f16(
            false, aF[mt].v, false, bF[nt2].v,
            (short)0, acc[mt][nt2].v, false, false);
    __syncthreads();                  // done reading buf bf; next-next issue may overwrite it
  }

  // Store C per 32-bit C/D layout: lanes 0-15 -> M=i, lanes 16-31 -> M=i+8
  #pragma unroll
  for (int mt = 0; mt < 2; ++mt) {
    #pragma unroll
    for (int nt2 = 0; nt2 < 2; ++nt2) {
      int n = n0 + wc * 32 + nt2 * 16 + lh;
      if (n < Nv) {
        int mb = m0 + wr * 32 + mt * 16 + (hi ? 8 : 0);
        #pragma unroll
        for (int i = 0; i < 8; ++i)
          C[(size_t)(mb + i) * Nv + n] = acc[mt][nt2].f[i];
      }
    }
  }
}

// ---------------------------------------------------------------------------
// Row-wise layernorm in place:  X[row,:] = LN(X[row,:] + bias) * g + be
// ---------------------------------------------------------------------------
__global__ __launch_bounds__(256) void layernorm_rows(
    float* __restrict__ X, const float* __restrict__ bias,
    const float* __restrict__ g, const float* __restrict__ be, int H)
{
  __shared__ float r1[256], r2[256];
  const int tid = threadIdx.x;
  float* row = X + (size_t)blockIdx.x * H;
  float s = 0.f, s2 = 0.f;
  for (int h = tid; h < H; h += 256) {
    float v = row[h] + bias[h];
    s += v; s2 += v * v;
  }
  r1[tid] = s; r2[tid] = s2; __syncthreads();
  for (int off = 128; off > 0; off >>= 1) {
    if (tid < off) { r1[tid] += r1[tid + off]; r2[tid] += r2[tid + off]; }
    __syncthreads();
  }
  float mu   = r1[0] / H;
  float var  = r2[0] / H - mu * mu;
  float rstd = rsqrtf(var + 1e-5f);
  for (int h = tid; h < H; h += 256) {
    float v = row[h] + bias[h];
    row[h] = (v - mu) * rstd * g[h] + be[h];
  }
}

// ---------------------------------------------------------------------------
// Per-timestep fused: layernorm(hc_pre + b_h) -> LIF update -> spikes
// One block per batch row b.
// ---------------------------------------------------------------------------
__global__ __launch_bounds__(256) void rnn_update(
    const float* __restrict__ hc_pre,   // [B, H]
    const float* __restrict__ ic,       // [B*T, H]
    const float* __restrict__ b_h, const float* __restrict__ g_h,
    const float* __restrict__ be_h, const float* __restrict__ tau_m,
    const float* __restrict__ tau_n,
    float* __restrict__ d_in, float* __restrict__ mem,     // state [B,H]
    float* __restrict__ mems_out, float* __restrict__ spks_out,
    _Float16* __restrict__ spk_h,       // [B, KP_H] next-step GEMM input
    int t)
{
  __shared__ float r1[256], r2[256];
  const int tid = threadIdx.x;
  const int b   = blockIdx.x;
  const float* hp = hc_pre + (size_t)b * HH;

  float s = 0.f, s2 = 0.f;
  for (int h = tid; h < HH; h += 256) {
    float v = hp[h] + b_h[h];
    s += v; s2 += v * v;
  }
  r1[tid] = s; r2[tid] = s2; __syncthreads();
  for (int off = 128; off > 0; off >>= 1) {
    if (tid < off) { r1[tid] += r1[tid + off]; r2[tid] += r2[tid + off]; }
    __syncthreads();
  }
  float mu   = r1[0] / HH;
  float var  = r2[0] / HH - mu * mu;
  float rstd = rsqrtf(var + 1e-5f);

  const size_t rowo = ((size_t)b * TT + t) * HH;
  const size_t bo   = (size_t)b * HH;
  for (int h = tid; h < HH; h += 256) {
    float v    = hp[h] + b_h[h];
    float hc   = (v - mu) * rstd * g_h[h] + be_h[h];
    float comb = ic[rowo + h] + hc;
    float sn   = 1.f / (1.f + __expf(-tau_n[h]));       // sigmoid(tau_n)
    float al   = __expf(-1.f / fmaxf(tau_m[h], 0.01f)); // alpha
    float di   = sn * d_in[bo + h] + (1.f - sn) * comb;
    float m    = al * mem[bo + h] + (1.f - al) * di;
    float sp   = (m - 0.5f) > 0.f ? 1.f : 0.f;          // VTH = 0.5
    d_in[bo + h] = di;
    mem[bo + h]  = m;
    mems_out[rowo + h] = m;
    spks_out[rowo + h] = sp;
    spk_h[(size_t)b * KP_H + h] = (_Float16)sp;
  }
}

// ---------------------------------------------------------------------------
// fp32 -> f16 with zero-padding of both K (cols) and N (rows) dimensions
// ---------------------------------------------------------------------------
__global__ void convert_pad(const float* __restrict__ src,
                            _Float16* __restrict__ dst,
                            int rows, int cols, int colsPad, long total)
{
  long i = (long)blockIdx.x * blockDim.x + threadIdx.x;
  if (i >= total) return;
  int r = (int)(i / colsPad), c = (int)(i % colsPad);
  dst[i] = (r < rows && c < cols) ? (_Float16)src[(size_t)r * cols + c]
                                  : (_Float16)0.f;
}

__global__ void init_state(const float* __restrict__ mem_init,
                           float* __restrict__ d_in, float* __restrict__ mem,
                           _Float16* __restrict__ spk_h)
{
  long i = (long)blockIdx.x * blockDim.x + threadIdx.x;
  if (i < (long)BB * HH) { d_in[i] = 0.f; mem[i] = mem_init[i]; }
  if (i < (long)BB * KP_H) spk_h[i] = (_Float16)0.f;
}

// ---------------------------------------------------------------------------
extern "C" void kernel_launch(void* const* d_in, const int* in_sizes, int n_in,
                              void* d_out, int out_size, void* d_ws, size_t ws_size,
                              hipStream_t stream)
{
  (void)in_sizes; (void)n_in; (void)out_size; (void)ws_size;

  const float* x        = (const float*)d_in[0];
  const float* mem_init = (const float*)d_in[1];
  const float* W_in     = (const float*)d_in[2];
  const float* b_in     = (const float*)d_in[3];
  const float* g_in     = (const float*)d_in[4];
  const float* be_in    = (const float*)d_in[5];
  const float* W_h      = (const float*)d_in[6];
  const float* b_h      = (const float*)d_in[7];
  const float* g_h      = (const float*)d_in[8];
  const float* be_h     = (const float*)d_in[9];
  const float* tau_m    = (const float*)d_in[10];
  const float* tau_n    = (const float*)d_in[11];

  float* out      = (float*)d_out;
  float* mems_out = out;
  float* spks_out = out + (size_t)BB * TT * HH;

  // Workspace carve-out (256B-aligned slabs)
  char* ws = (char*)d_ws;
  size_t off = 0;
  auto carve = [&](size_t bytes) {
    void* p = ws + off;
    off = (off + bytes + 255) & ~(size_t)255;
    return p;
  };
  _Float16* xh   = (_Float16*)carve((size_t)BB * TT * KP_IN * 2); // 60.6 MB
  _Float16* Winh = (_Float16*)carve((size_t)NP * KP_IN * 2);      //  3.6 MB
  _Float16* Whh  = (_Float16*)carve((size_t)NP * KP_H * 2);       //  4.6 MB (L2-resident)
  float*    ic   = (float*)   carve((size_t)BB * TT * HH * 4);    // 153.6 MB
  float*    hcp  = (float*)   carve((size_t)BB * HH * 4);
  float*    dinw = (float*)   carve((size_t)BB * HH * 4);
  float*    memw = (float*)   carve((size_t)BB * HH * 4);
  _Float16* spkh = (_Float16*)carve((size_t)BB * KP_H * 2);

  // 1) f16 conversion + padding (spikes are exact in f16; weights keep 10 mantissa bits)
  long tx = (long)BB * TT * KP_IN;
  convert_pad<<<(tx + 255) / 256, 256, 0, stream>>>(x, xh, BB * TT, DD, KP_IN, tx);
  long tw = (long)NP * KP_IN;
  convert_pad<<<(tw + 255) / 256, 256, 0, stream>>>(W_in, Winh, HH, DD, KP_IN, tw);
  long th = (long)NP * KP_H;
  convert_pad<<<(th + 255) / 256, 256, 0, stream>>>(W_h, Whh, HH, HH, KP_H, th);
  init_state<<<(BB * KP_H + 255) / 256, 256, 0, stream>>>(mem_init, dinw, memw, spkh);

  // 2) Batched input GEMM over all timesteps + layernorm
  dim3 g1(NP / 64, (BB * TT) / 128);          // 24 x 200
  wmma_gemm_f16<<<g1, 256, 0, stream>>>(xh, Winh, ic, BB * TT, HH, KP_IN);
  layernorm_rows<<<BB * TT, 256, 0, stream>>>(ic, b_in, g_in, be_in, HH);

  // 3) Sequential recurrent steps: spk GEMM (W_h stays hot in L2) + fused update
  dim3 g2(NP / 64, BB / 128);                 // 24 x 4
  for (int t = 0; t < TT; ++t) {
    wmma_gemm_f16<<<g2, 256, 0, stream>>>(spkh, Whh, hcp, BB, HH, KP_H);
    rnn_update<<<BB, 256, 0, stream>>>(hcp, ic, b_h, g_h, be_h, tau_m, tau_n,
                                       dinw, memw, mems_out, spks_out, spkh, t);
  }
}
